// Classifier_5179730559451
// MI455X (gfx1250) — matrix operand order repaired and checked by reading, not compiled, not run on hardware
//
#include <hip/hip_runtime.h>
#include <math.h>
#include <stdint.h>

typedef __attribute__((ext_vector_type(16))) _Float16 v16h;
typedef __attribute__((ext_vector_type(8)))  float    v8f;

#define DEV_INLINE __device__ __forceinline__

namespace {

constexpr int Bsz = 8, Tt = 25, CIN = 5, DDi = 5, HHi = 6, WWi = 6;
constexpr int BT = Bsz * Tt;            // 200
constexpr int GRUH = 12, CNN_OUT = 256;
constexpr int ND = 301;                 // NODE_DIM
constexpr int NDW = (ND + 31) / 32;     // 10 dword-lanes per row
constexpr int Nn = Bsz * 4096;          // 32768
constexpr int Ee = Nn * 16;             // 524288
constexpr float EPSF = 1e-5f;

// ======================= fills =======================
__global__ void fill_u32(unsigned* p, unsigned v, int n) {
  int i = blockIdx.x * blockDim.x + threadIdx.x;
  if (i < n) p[i] = v;
}

// ======================= direct 3D conv (3x3x3, pad 1, stride 1) ===========
__global__ void conv3d_k(const float* __restrict__ in, const float* __restrict__ w,
                         const float* __restrict__ bias, float* __restrict__ out,
                         int NB, int CI, int CO, int D, int H, int W) {
  int idx = blockIdx.x * blockDim.x + threadIdx.x;
  int S = D * H * W;
  int total = NB * CO * S;
  if (idx >= total) return;
  int s  = idx % S;
  int co = (idx / S) % CO;
  int nb = idx / (S * CO);
  int ww = s % W, hh = (s / W) % H, dd = s / (W * H);
  float acc = bias[co];
  const float* wp = w + (size_t)co * CI * 27;
  for (int ci = 0; ci < CI; ++ci) {
    const float* ip = in + ((size_t)nb * CI + ci) * S;
    const float* wc = wp + ci * 27;
    for (int kd = 0; kd < 3; ++kd) {
      int id = dd + kd - 1;
      if ((unsigned)id >= (unsigned)D) continue;
      for (int kh = 0; kh < 3; ++kh) {
        int ih = hh + kh - 1;
        if ((unsigned)ih >= (unsigned)H) continue;
        for (int kw = 0; kw < 3; ++kw) {
          int iw = ww + kw - 1;
          if ((unsigned)iw >= (unsigned)W) continue;
          acc += ip[(id * H + ih) * W + iw] * wc[(kd * 3 + kh) * 3 + kw];
        }
      }
    }
  }
  out[idx] = acc;
}

// ======================= 2x2x2 max pool, stride 2, -inf pad ================
__global__ void maxpool_k(const float* __restrict__ in, float* __restrict__ out,
                          int NB, int C, int Di, int Hi, int Wi,
                          int Do, int Ho, int Wo, int padD, int padHW) {
  int idx = blockIdx.x * blockDim.x + threadIdx.x;
  int So = Do * Ho * Wo;
  int total = NB * C * So;
  if (idx >= total) return;
  int s = idx % So;
  int c = (idx / So) % C;
  int nb = idx / (So * C);
  int ow = s % Wo, oh = (s / Wo) % Ho, od = s / (Wo * Ho);
  const float* ip = in + ((size_t)nb * C + c) * (Di * Hi * Wi);
  float m = -INFINITY;
  for (int kd = 0; kd < 2; ++kd) {
    int id = od * 2 - padD + kd;
    if ((unsigned)id >= (unsigned)Di) continue;
    for (int kh = 0; kh < 2; ++kh) {
      int ih = oh * 2 - padHW + kh;
      if ((unsigned)ih >= (unsigned)Hi) continue;
      for (int kw = 0; kw < 2; ++kw) {
        int iw = ow * 2 - padHW + kw;
        if ((unsigned)iw >= (unsigned)Wi) continue;
        m = fmaxf(m, ip[(id * Hi + ih) * Wi + iw]);
      }
    }
  }
  out[idx] = m;
}

// ======================= train-mode batch norm =============================
__global__ void bn_stats_k(const float* __restrict__ x, int R, int C, int S,
                           float* __restrict__ mean, float* __restrict__ inv) {
  __shared__ float ssum[256], ssq[256];
  int c = blockIdx.x, tid = threadIdx.x;
  float sum = 0.f, sq = 0.f;
  int tot = R * S;
  for (int i = tid; i < tot; i += 256) {
    int r = i / S, s = i % S;
    float v = x[((size_t)r * C + c) * S + s];
    sum += v; sq += v * v;
  }
  ssum[tid] = sum; ssq[tid] = sq;
  __syncthreads();
  for (int off = 128; off; off >>= 1) {
    if (tid < off) { ssum[tid] += ssum[tid + off]; ssq[tid] += ssq[tid + off]; }
    __syncthreads();
  }
  if (tid == 0) {
    float m = ssum[0] / (float)tot;
    float v = ssq[0] / (float)tot - m * m;
    mean[c] = m;
    inv[c] = rsqrtf(fmaxf(v, 0.f) + EPSF);
  }
}
__global__ void bn_apply_k(float* __restrict__ x, const float* __restrict__ mean,
                           const float* __restrict__ inv, const float* __restrict__ g,
                           const float* __restrict__ b, int C, int S, int total, int relu) {
  int i = blockIdx.x * blockDim.x + threadIdx.x;
  if (i >= total) return;
  int c = (i / S) % C;
  float y = (x[i] - mean[c]) * inv[c] * g[c] + b[c];
  if (relu) y = fmaxf(y, 0.f);
  x[i] = y;
}

// ======================= f32 -> padded f16 conversion ======================
__global__ void cvt_f16_pad_k(const float* __restrict__ src, _Float16* __restrict__ dst,
                              int R, int K, int Kp, int total) {
  int i = blockIdx.x * blockDim.x + threadIdx.x;
  if (i >= total) return;
  int r = i / Kp, k = i % Kp;
  dst[i] = (r < R && k < K) ? (_Float16)src[(size_t)r * K + k] : (_Float16)0.f;
}

// ======================= async LDS copy helpers (CDNA5) ====================
DEV_INLINE void async_cp16(unsigned lds_off, const _Float16* g) {
  // GLOBAL_LOAD_ASYNC_TO_LDS_B128: 16B global -> LDS, tracked by ASYNCcnt
  asm volatile("global_load_async_to_lds_b128 %0, %1, off"
               :: "v"(lds_off), "v"(g) : "memory");
}
DEV_INLINE void wait_async0() {
  asm volatile("s_wait_asynccnt 0x0" ::: "memory");
}

// ======================= WMMA GEMM =========================================
// C[M,N] = A16[Mp,Kp] * B16[Np,Kp]^T (+bias). A16/B16 are zero-padded f16,
// so every tile load is unconditional. Tiles staged with async LDS copies
// (double buffered): copy of tile k+1 overlaps the WMMAs of tile k.
// Block tile 64x64, 8 wave32 waves, each wave = 32x16 (two f32 accumulators).
__global__ __launch_bounds__(256)
void gemm_wmma_f16(const _Float16* __restrict__ A, const _Float16* __restrict__ B,
                   const float* __restrict__ bias, float* __restrict__ C,
                   int M, int N, int Kp) {
  __shared__ __align__(32) _Float16 smem[2][2][64 * 32];  // [buf][A|B][tile]
  const int tid  = threadIdx.x;
  const int lane = tid & 31;
  const int wave = tid >> 5;
  const int bm0 = blockIdx.y * 64;
  const int bn0 = blockIdx.x * 64;
  const int wm = (wave >> 2) * 32;
  const int wn = (wave & 3) * 16;
  const int half_hi = (lane >= 16);
  const int fr = lane & 15;

  const int lrow = tid >> 2;          // 0..63
  const int lcol = (tid & 3) * 8;     // 0,8,16,24
  const unsigned lds0 = (unsigned)(uintptr_t)(&smem[0][0][0]);
  const unsigned ldsThread = (unsigned)(lrow * 32 + lcol) * 2u;
  const _Float16* gA = A + (size_t)(bm0 + lrow) * Kp + lcol;
  const _Float16* gB = B + (size_t)(bn0 + lrow) * Kp + lcol;

  auto stage = [&](int buf, int k0) {
    unsigned dA = lds0 + (unsigned)buf * 8192u + ldsThread;
    async_cp16(dA, gA + k0);
    async_cp16(dA + 4096u, gB + k0);
  };

  v8f acc0 = {}; v8f acc1 = {};
  const int nk = Kp >> 5;
  stage(0, 0);
  for (int kt = 0; kt < nk; ++kt) {
    wait_async0();
    __syncthreads();
    if (kt + 1 < nk) stage((kt + 1) & 1, (kt + 1) << 5);

    const _Float16* sa = &smem[kt & 1][0][0];
    const _Float16* sb = &smem[kt & 1][1][0];
    // B fragment: col n = wn+fr; halves 0..7 -> K=(hi?8:0)+0..7, halves 8..15 -> +16
    union { uint4 q[2]; v16h v; } bf, af0, af1;
    const _Float16* pb = sb + (wn + fr) * 32 + (half_hi ? 8 : 0);
    bf.q[0] = *(const uint4*)pb;
    bf.q[1] = *(const uint4*)(pb + 16);
    // A fragments: halves 0..7 -> K=(hi?8:0)+0..7, halves 8..15 -> +16
    const _Float16* pa0 = sa + (wm + fr) * 32 + (half_hi ? 8 : 0);
    af0.q[0] = *(const uint4*)pa0;
    af0.q[1] = *(const uint4*)(pa0 + 16);
    const _Float16* pa1 = pa0 + 16 * 32;
    af1.q[0] = *(const uint4*)pa1;
    af1.q[1] = *(const uint4*)(pa1 + 16);

    acc0 = __builtin_amdgcn_wmma_f32_16x16x32_f16(false, af0.v, false, bf.v,
                                                  (short)0, acc0, false, false);
    acc1 = __builtin_amdgcn_wmma_f32_16x16x32_f16(false, af1.v, false, bf.v,
                                                  (short)0, acc1, false, false);
  }

  // store: VGPR r, lane l -> M = r + (l<16?0:8), N = l%16
  int ccol = bn0 + wn + fr;
  if (ccol < N) {
    float bv = bias ? bias[ccol] : 0.f;
    #pragma unroll
    for (int r = 0; r < 8; ++r) {
      int row0 = bm0 + wm + r + (half_hi ? 8 : 0);
      if (row0 < M) C[(size_t)row0 * N + ccol] = acc0[r] + bv;
      int row1 = row0 + 16;
      if (row1 < M) C[(size_t)row1 * N + ccol] = acc1[r] + bv;
    }
  }
}

// ======================= GRU layer (torch gate order r,z,n) ================
__global__ void gru_k(const float* __restrict__ x, int IN,
                      const float* __restrict__ Wih, const float* __restrict__ bih,
                      const float* __restrict__ Whh, const float* __restrict__ bhh,
                      float* __restrict__ out) {
  __shared__ float h[8][12];
  __shared__ float xs[8 * 256];
  int b = threadIdx.x / 12, j = threadIdx.x % 12;
  h[b][j] = 0.f;
  __syncthreads();
  for (int t = 0; t < Tt; ++t) {
    for (int i = threadIdx.x; i < 8 * IN; i += 96) {
      int bb = i / IN, ii = i % IN;
      xs[bb * IN + ii] = x[((size_t)(bb * Tt + t)) * IN + ii];
    }
    __syncthreads();
    float xr = bih[j], xz = bih[12 + j], xn = bih[24 + j];
    for (int i = 0; i < IN; ++i) {
      float xv = xs[b * IN + i];
      xr += Wih[(size_t)j * IN + i] * xv;
      xz += Wih[(size_t)(12 + j) * IN + i] * xv;
      xn += Wih[(size_t)(24 + j) * IN + i] * xv;
    }
    float hr = bhh[j], hz = bhh[12 + j], hn = bhh[24 + j];
    for (int k = 0; k < 12; ++k) {
      float hv = h[b][k];
      hr += Whh[j * 12 + k] * hv;
      hz += Whh[(12 + j) * 12 + k] * hv;
      hn += Whh[(24 + j) * 12 + k] * hv;
    }
    float r = 1.f / (1.f + __expf(-(xr + hr)));
    float z = 1.f / (1.f + __expf(-(xz + hz)));
    float n = tanhf(xn + r * hn);
    float hnew = (1.f - z) * n + z * h[b][j];
    __syncthreads();
    h[b][j] = hnew;
    out[((size_t)(b * Tt + t)) * 12 + j] = hnew;
    __syncthreads();
  }
}

// ======================= node feature build ================================
__global__ void build_nodes_k(const float* __restrict__ enc, float* __restrict__ X) {
  int i = blockIdx.x * blockDim.x + threadIdx.x;
  if (i >= Nn * ND) return;
  int n = i / ND, c = i % ND;
  X[i] = (c == 0) ? 0.f : enc[(n >> 12) * 300 + (c - 1)];
}

// ======================= CSR by dst (built once) ===========================
__global__ void deg_k(const int* __restrict__ dst, int* __restrict__ deg, int E) {
  int e = blockIdx.x * blockDim.x + threadIdx.x;
  if (e < E) atomicAdd(&deg[dst[e]], 1);
}
// single block exclusive scan (n == chunk*256)
__global__ void scan_k(const int* __restrict__ deg, int* __restrict__ rowptr, int n) {
  __shared__ int part[256];
  int tid = threadIdx.x;
  int chunk = (n + 255) / 256;
  int begin = tid * chunk, end = min(begin + chunk, n);
  int s = 0;
  for (int i = begin; i < end; ++i) s += deg[i];
  part[tid] = s;
  __syncthreads();
  if (tid == 0) {
    int acc = 0;
    for (int i = 0; i < 256; ++i) { int v = part[i]; part[i] = acc; acc += v; }
  }
  __syncthreads();
  int acc = part[tid];
  for (int i = begin; i < end; ++i) { rowptr[i] = acc; acc += deg[i]; }
  if (end == n) rowptr[n] = acc;
}
__global__ void copy_i32_k(const int* __restrict__ s, int* __restrict__ d, int n) {
  int i = blockIdx.x * blockDim.x + threadIdx.x;
  if (i < n) d[i] = s[i];
}
__global__ void csr_fill_k(const int* __restrict__ dst, int* __restrict__ cursor,
                           int* __restrict__ eidx, int E) {
  int e = blockIdx.x * blockDim.x + threadIdx.x;
  if (e >= E) return;
  int slot = atomicAdd(&cursor[dst[e]], 1);
  eidx[slot] = e;
}

// ======================= GAT edge logits (wave per edge) ===================
__global__ void gat_logits_k(const float* __restrict__ xl, const float* __restrict__ xr,
                             const int* __restrict__ src, const int* __restrict__ dst,
                             const float* __restrict__ ea, const float* __restrict__ We,
                             const float* __restrict__ att, float* __restrict__ logits,
                             int E) {
  int gid = blockIdx.x * blockDim.x + threadIdx.x;
  int e = gid >> 5, lane = gid & 31;
  if (e >= E) return;
  int s = src[e], d = dst[e];
  float a = ea[e];
  float acc = 0.f;
  for (int o = lane; o < ND; o += 32) {
    float m = xl[(size_t)s * ND + o] + xr[(size_t)d * ND + o] + a * We[o];
    m = (m > 0.f) ? m : 0.2f * m;
    acc += att[o] * m;
  }
  for (int off = 16; off; off >>= 1) acc += __shfl_xor(acc, off, 32);
  if (lane == 0) logits[e] = acc;
}

// ============== fused edge-softmax + mean aggregation (wave per node) ======
__global__ void gat_softmax_agg_k(const float* __restrict__ xl,
                                  const float* __restrict__ logits,
                                  const int* __restrict__ src,
                                  const int* __restrict__ rowptr,
                                  const int* __restrict__ eidx,
                                  const float* __restrict__ bias,
                                  float* __restrict__ out, int N) {
  int gid = blockIdx.x * blockDim.x + threadIdx.x;
  int n = gid >> 5, lane = gid & 31;
  if (n >= N) return;
  int beg = rowptr[n], end = rowptr[n + 1];
  float mx = -INFINITY;
  for (int i = beg + lane; i < end; i += 32) mx = fmaxf(mx, logits[eidx[i]]);
  for (int off = 16; off; off >>= 1) mx = fmaxf(mx, __shfl_xor(mx, off, 32));
  float den = 0.f;
  for (int i = beg + lane; i < end; i += 32) den += __expf(logits[eidx[i]] - mx);
  for (int off = 16; off; off >>= 1) den += __shfl_xor(den, off, 32);
  float inv_den = (end > beg) ? 1.f / den : 0.f;
  float invc = 1.f / fmaxf((float)(end - beg), 1.f);

  float acc[NDW];
  #pragma unroll
  for (int u = 0; u < NDW; ++u) acc[u] = 0.f;
  for (int i = beg; i < end; ++i) {
    int e = eidx[i];
    float alpha = __expf(logits[e] - mx) * inv_den;
    const float* xs = xl + (size_t)src[e] * ND;
    #pragma unroll
    for (int u = 0; u < NDW; ++u) {
      int o = lane + u * 32;
      if (o < ND) acc[u] += xs[o] * alpha;
    }
  }
  #pragma unroll
  for (int u = 0; u < NDW; ++u) {
    int o = lane + u * 32;
    if (o < ND) out[(size_t)n * ND + o] = acc[u] * invc + bias[o];
  }
}

// ======================= GAT3 (output dim 1) ===============================
__global__ void gat3_proj_k(const float* __restrict__ X,
                            const float* __restrict__ Wl, const float* __restrict__ bl,
                            const float* __restrict__ Wr, const float* __restrict__ br,
                            float* __restrict__ xl1, float* __restrict__ xr1, int N) {
  int gid = blockIdx.x * blockDim.x + threadIdx.x;
  int n = gid >> 5, lane = gid & 31;
  if (n >= N) return;
  float al = 0.f, ar = 0.f;
  for (int k = lane; k < ND; k += 32) {
    float v = X[(size_t)n * ND + k];
    al += v * Wl[k];
    ar += v * Wr[k];
  }
  for (int off = 16; off; off >>= 1) {
    al += __shfl_xor(al, off, 32);
    ar += __shfl_xor(ar, off, 32);
  }
  if (lane == 0) { xl1[n] = al + bl[0]; xr1[n] = ar + br[0]; }
}

__global__ void gat3_logits_k(const float* __restrict__ xl1, const float* __restrict__ xr1,
                              const int* __restrict__ src, const int* __restrict__ dst,
                              const float* __restrict__ ea, const float* __restrict__ We,
                              const float* __restrict__ att, float* __restrict__ logits,
                              int E) {
  int e = blockIdx.x * blockDim.x + threadIdx.x;
  if (e >= E) return;
  float m = xl1[src[e]] + xr1[dst[e]] + ea[e] * We[0];
  m = (m > 0.f) ? m : 0.2f * m;
  logits[e] = att[0] * m;
}

__global__ void gat3_out_k(const float* __restrict__ xl1, const float* __restrict__ logits,
                           const int* __restrict__ src, const int* __restrict__ rowptr,
                           const int* __restrict__ eidx, const float* __restrict__ bias,
                           const unsigned char* __restrict__ mask, const float* __restrict__ y,
                           float* __restrict__ out, int N) {
  int n = blockIdx.x * blockDim.x + threadIdx.x;
  if (n >= N) return;
  int beg = rowptr[n], end = rowptr[n + 1];
  float m = -INFINITY;
  for (int i = beg; i < end; ++i) m = fmaxf(m, logits[eidx[i]]);
  float den = 0.f, s = 0.f;
  for (int i = beg; i < end; ++i) {
    int e = eidx[i];
    float w = __expf(logits[e] - m);
    den += w;
    s += w * xl1[src[e]];
  }
  float mean = (den > 0.f) ? (s / den) : 0.f;
  float g = mean / fmaxf((float)(end - beg), 1.f) + bias[0];
  float yp = 1.f / (1.f + __expf(-g));
  bool mk = mask[n] != 0;
  out[n] = mk ? yp : 0.f;
  out[N + n] = mk ? y[n] : 0.f;
}

} // namespace

// ============================ host launcher ================================
extern "C" void kernel_launch(void* const* d_in, const int* in_sizes, int n_in,
                              void* d_out, int out_size, void* d_ws, size_t ws_size,
                              hipStream_t stream) {
  (void)in_sizes; (void)n_in; (void)out_size; (void)ws_size;

  // ---- inputs (setup_inputs() insertion order, params depth-first) ----
  const float* X_batch   = (const float*)d_in[0];
  const int*   ei        = (const int*)d_in[1];
  const float* edge_attr = (const float*)d_in[2];
  const unsigned char* train_mask = (const unsigned char*)d_in[3];
  const float* yv        = (const float*)d_in[4];
  auto P = [&](int i) { return (const float*)d_in[i]; };
  const float *conv1_w = P(5),  *conv1_b = P(6),  *bn1_g = P(7),  *bn1_b = P(8);
  const float *conv2_w = P(9),  *conv2_b = P(10), *bn2_g = P(11), *bn2_b = P(12);
  const float *conv3_w = P(13), *conv3_b = P(14), *bn3_g = P(15), *bn3_b = P(16);
  const float *fc1_w = P(17), *fc1_b = P(18), *bn4_g = P(19), *bn4_b = P(20);
  const float *fc2_w = P(21), *fc2_b = P(22), *bn5_g = P(23), *bn5_b = P(24);
  const float *g0Wih = P(25), *g0Whh = P(26), *g0bih = P(27), *g0bhh = P(28);
  const float *g1Wih = P(29), *g1Whh = P(30), *g1bih = P(31), *g1bhh = P(32);
  const float *bng1_g = P(33), *bng1_b = P(34);
  const float *bng2_g = P(35), *bng2_b = P(36);
  const float *bng3_g = P(37), *bng3_b = P(38);
  const float *ga1Wl = P(39), *ga1bl = P(40), *ga1Wr = P(41), *ga1br = P(42),
              *ga1We = P(43), *ga1att = P(44), *ga1bias = P(45);
  const float *ga2Wl = P(46), *ga2bl = P(47), *ga2Wr = P(48), *ga2br = P(49),
              *ga2We = P(50), *ga2att = P(51), *ga2bias = P(52);
  const float *ga3Wl = P(53), *ga3bl = P(54), *ga3Wr = P(55), *ga3br = P(56),
              *ga3We = P(57), *ga3att = P(58), *ga3bias = P(59);

  const int* src = ei;
  const int* dst = ei + Ee;
  float* out = (float*)d_out;

  // ---- workspace carve ----
  char* base = (char*)d_ws;
  size_t off = 0;
  auto alloc = [&](size_t bytes) -> void* {
    off = (off + 255) & ~(size_t)255;
    void* p = base + off;
    off += bytes;
    return p;
  };
  float* c1  = (float*)alloc((size_t)BT * 64 * 180 * 4);
  float* c2  = (float*)alloc((size_t)BT * 64 * 180 * 4);
  float* p1  = (float*)alloc((size_t)BT * 64 * 48 * 4);
  float* c3  = (float*)alloc((size_t)BT * 256 * 48 * 4);
  float* p2  = (float*)alloc((size_t)BT * 2048 * 4);
  float* f1  = (float*)alloc((size_t)BT * 512 * 4);
  float* f2  = (float*)alloc((size_t)BT * 256 * 4);
  float* g1o = (float*)alloc((size_t)BT * GRUH * 4);
  float* enc = (float*)alloc((size_t)BT * GRUH * 4);
  float* meanb = (float*)alloc(2048 * 4);
  float* invb  = (float*)alloc(2048 * 4);
  float* XA = (float*)alloc((size_t)Nn * ND * 4);   // features (layer input/output)
  float* XB = (float*)alloc((size_t)Nn * ND * 4);   // xl
  float* XC = (float*)alloc((size_t)Nn * ND * 4);   // xr
  _Float16* A16 = (_Float16*)alloc((size_t)Nn * 320 * 2);      // padded f16 A
  _Float16* B16 = (_Float16*)alloc((size_t)512 * 2048 * 2);    // padded f16 B
  float* logits = (float*)alloc((size_t)Ee * 4);
  int* deg    = (int*)alloc((size_t)Nn * 4);
  int* rowptr = (int*)alloc((size_t)(Nn + 1) * 4);
  int* cursor = (int*)alloc((size_t)Nn * 4);
  int* eidx   = (int*)alloc((size_t)Ee * 4);
  float* xl1  = (float*)alloc((size_t)Nn * 4);
  float* xr1  = (float*)alloc((size_t)Nn * 4);

  auto blks = [](long n) { return (int)((n + 255) / 256); };

  // convert f32 [R,K] -> zero-padded f16 [Rp,Kp]
  auto cvt = [&](const float* srcp, _Float16* dstp, int R, int K, int Rp, int Kp) {
    int total = Rp * Kp;
    cvt_f16_pad_k<<<blks(total), 256, 0, stream>>>(srcp, dstp, R, K, Kp, total);
  };
  // C[M,N] = A[M,K] * W[N,K]^T + bias (A already converted into A16 w/ Kp)
  auto gemm_pre = [&](const _Float16* A, const float* Wt, const float* bias, float* Cc,
                      int M, int N, int K) {
    int Kp = (K + 31) & ~31;
    int Mp = (M + 63) & ~63;
    int Np = (N + 63) & ~63;
    cvt(Wt, B16, N, K, Np, Kp);
    dim3 g(Np / 64, Mp / 64);
    gemm_wmma_f16<<<g, 256, 0, stream>>>(A, B16, bias, Cc, M, N, Kp);
  };
  auto gemm = [&](const float* A, const float* Wt, const float* bias, float* Cc,
                  int M, int N, int K) {
    int Kp = (K + 31) & ~31;
    int Mp = (M + 63) & ~63;
    cvt(A, A16, M, K, Mp, Kp);
    gemm_pre(A16, Wt, bias, Cc, M, N, K);
  };
  auto bn = [&](float* x, int R, int C, int S, const float* gg, const float* bb, int relu) {
    bn_stats_k<<<C, 256, 0, stream>>>(x, R, C, S, meanb, invb);
    long tot = (long)R * C * S;
    bn_apply_k<<<blks(tot), 256, 0, stream>>>(x, meanb, invb, gg, bb, C, S, (int)tot, relu);
  };

  // ================= CNN =================
  conv3d_k<<<blks((long)BT * 64 * 180), 256, 0, stream>>>(X_batch, conv1_w, conv1_b, c1,
                                                          BT, CIN, 64, DDi, HHi, WWi);
  bn(c1, BT, 64, 180, bn1_g, bn1_b, 1);
  conv3d_k<<<blks((long)BT * 64 * 180), 256, 0, stream>>>(c1, conv2_w, conv2_b, c2,
                                                          BT, 64, 64, DDi, HHi, WWi);
  bn(c2, BT, 64, 180, bn2_g, bn2_b, 1);
  maxpool_k<<<blks((long)BT * 64 * 48), 256, 0, stream>>>(c2, p1, BT, 64, 5, 6, 6,
                                                          3, 4, 4, 1, 1);
  conv3d_k<<<blks((long)BT * 256 * 48), 256, 0, stream>>>(p1, conv3_w, conv3_b, c3,
                                                          BT, 64, 256, 3, 4, 4);
  bn(c3, BT, 256, 48, bn3_g, bn3_b, 1);
  maxpool_k<<<blks((long)BT * 256 * 8), 256, 0, stream>>>(c3, p2, BT, 256, 3, 4, 4,
                                                          2, 2, 2, 1, 0);
  // ================= FC (WMMA) =================
  gemm(p2, fc1_w, fc1_b, f1, BT, 512, 2048);
  bn(f1, BT, 512, 1, bn4_g, bn4_b, 1);
  gemm(f1, fc2_w, fc2_b, f2, BT, 256, 512);
  bn(f2, BT, 256, 1, bn5_g, bn5_b, 1);

  // ================= GRU =================
  gru_k<<<1, 96, 0, stream>>>(f2, CNN_OUT, g0Wih, g0bih, g0Whh, g0bhh, g1o);
  gru_k<<<1, 96, 0, stream>>>(g1o, GRUH, g1Wih, g1bih, g1Whh, g1bhh, enc);

  // ================= node features + BN-g1 =================
  build_nodes_k<<<blks((long)Nn * ND), 256, 0, stream>>>(enc, XA);
  bn(XA, Nn, ND, 1, bng1_g, bng1_b, 0);

  // ================= CSR by dst (shared by all 3 GAT layers) ===============
  fill_u32<<<blks(Nn), 256, 0, stream>>>((unsigned*)deg, 0u, Nn);
  deg_k<<<blks(Ee), 256, 0, stream>>>(dst, deg, Ee);
  scan_k<<<1, 256, 0, stream>>>(deg, rowptr, Nn);
  copy_i32_k<<<blks(Nn), 256, 0, stream>>>(rowptr, cursor, Nn);
  csr_fill_k<<<blks(Ee), 256, 0, stream>>>(dst, cursor, eidx, Ee);

  // ================= GATv2 layers 1 & 2 ====================================
  auto gat_layer = [&](const float* Wl, const float* bl, const float* Wr, const float* br,
                       const float* We, const float* att, const float* gbias) {
    cvt(XA, A16, Nn, ND, Nn, 320);                 // features once, reused for both GEMMs
    gemm_pre(A16, Wl, bl, XB, Nn, ND, ND);         // xl
    gemm_pre(A16, Wr, br, XC, Nn, ND, ND);         // xr
    gat_logits_k<<<blks((long)Ee * 32), 256, 0, stream>>>(XB, XC, src, dst, edge_attr,
                                                          We, att, logits, Ee);
    gat_softmax_agg_k<<<blks((long)Nn * 32), 256, 0, stream>>>(XB, logits, src, rowptr,
                                                               eidx, gbias, XA, Nn);
  };
  gat_layer(ga1Wl, ga1bl, ga1Wr, ga1br, ga1We, ga1att, ga1bias);
  bn(XA, Nn, ND, 1, bng2_g, bng2_b, 1);
  gat_layer(ga2Wl, ga2bl, ga2Wr, ga2br, ga2We, ga2att, ga2bias);
  bn(XA, Nn, ND, 1, bng3_g, bng3_b, 1);

  // ================= GAT3 (scalar output) =================
  gat3_proj_k<<<blks((long)Nn * 32), 256, 0, stream>>>(XA, ga3Wl, ga3bl, ga3Wr, ga3br,
                                                       xl1, xr1, Nn);
  gat3_logits_k<<<blks(Ee), 256, 0, stream>>>(xl1, xr1, src, dst, edge_attr,
                                              ga3We, ga3att, logits, Ee);
  gat3_out_k<<<blks(Nn), 256, 0, stream>>>(xl1, logits, src, rowptr, eidx, ga3bias,
                                           train_mask, yv, out, Nn);
}
